// MambaBlock_71416716198239
// MI455X (gfx1250) — matrix operand order repaired and checked by reading, not compiled, not run on hardware
//
#include <hip/hip_runtime.h>
#include <hip/hip_bf16.h>
#include <math.h>

typedef __attribute__((ext_vector_type(16))) __bf16 v16bf;
typedef __attribute__((ext_vector_type(8)))  __bf16 v8bf;
typedef __attribute__((ext_vector_type(8)))  float  v8f;

#define BM 128
#define BN 64
#define BK 32
#define ASTRIDE 40   // bf16 elems -> 80B row pitch (keeps 16B alignment, pads banks)

union FragU { v16bf v; v8bf h[2]; };

__device__ __forceinline__ v8bf pack8_bf16(const float* __restrict__ gp) {
    const float4 f0 = *(const float4*)(gp);
    const float4 f1 = *(const float4*)(gp + 4);
    v8bf p;
    p[0] = (__bf16)f0.x; p[1] = (__bf16)f0.y; p[2] = (__bf16)f0.z; p[3] = (__bf16)f0.w;
    p[4] = (__bf16)f1.x; p[5] = (__bf16)f1.y; p[6] = (__bf16)f1.z; p[7] = (__bf16)f1.w;
    return p;
}

// C[M,N] = act( A[M,K](row-major, stride lda) * W[N,K]^T (row-major, stride ldw) + bias )
// Requires M%128==0, N%64==0, K%32==0 (true for every call here).
// 256 threads = 8 waves; each wave owns a 2x2 grid of 16x16 WMMA tiles.
__global__ __launch_bounds__(256) void wmma_gemm_bf16(
    const float* __restrict__ A, int lda,
    const float* __restrict__ W, int ldw,
    float* __restrict__ C, int ldc,
    const float* __restrict__ bias,
    int M, int N, int K, int act)
{
    __shared__ __align__(16) __bf16 As[BM * ASTRIDE];   // 10 KB
    __shared__ __align__(16) __bf16 Ws[BN * ASTRIDE];   //  5 KB

    const int tid  = threadIdx.x;
    const int lane = tid & 31;
    const int wave = tid >> 5;                 // 0..7
    const int bm0  = blockIdx.y * BM;
    const int bn0  = blockIdx.x * BN;

    const int tr0 = (wave >> 1) * 2;           // tile rows {tr0, tr0+1} of 8
    const int tc0 = (wave & 1) * 2;            // tile cols {tc0, tc0+1} of 4
    const int laneM  = lane & 15;
    const int hiHalf = lane >> 4;

    // Cooperative-load assignments: 8 contiguous K-elems per chunk.
    // A: 128 rows x 4 chunks = 512 chunks -> 2 per thread. W: 256 chunks -> 1 per thread.
    const int ar0 = tid >> 2,        ac0 = (tid & 3) * 8;
    const int ar1 = (tid + 256) >> 2;          // = ar0 + 64
    const int wr  = tid >> 2,        wc  = (tid & 3) * 8;

    const float* gA0 = A + (size_t)(bm0 + ar0) * lda + ac0;
    const float* gA1 = A + (size_t)(bm0 + ar1) * lda + ac0;
    const float* gW  = W + (size_t)(bn0 + wr)  * ldw + wc;

    // LDS fragment read pointers (constant across K loop)
    const __bf16* arow0 = &As[(tr0 * 16 + laneM) * ASTRIDE];
    const __bf16* arow1 = arow0 + 16 * ASTRIDE;
    const __bf16* wrow0 = &Ws[((tc0 + 0) * 16 + laneM) * ASTRIDE] + hiHalf * 8;
    const __bf16* wrow1 = &Ws[((tc0 + 1) * 16 + laneM) * ASTRIDE] + hiHalf * 8;

    v8f acc00 = {}, acc01 = {}, acc10 = {}, acc11 = {};

    for (int k0 = 0; k0 < K; k0 += BK) {
        // Stage tiles: f32 -> bf16, packed b128 LDS stores
        *(v8bf*)&As[ar0 * ASTRIDE + ac0] = pack8_bf16(gA0 + k0);
        *(v8bf*)&As[ar1 * ASTRIDE + ac0] = pack8_bf16(gA1 + k0);
        *(v8bf*)&Ws[wr  * ASTRIDE + wc ] = pack8_bf16(gW  + k0);
        if (k0 + BK < K) {                    // prefetch next K tile (global_prefetch_b8)
            __builtin_prefetch(gA0 + k0 + BK, 0, 1);
            __builtin_prefetch(gA1 + k0 + BK, 0, 1);
            __builtin_prefetch(gW  + k0 + BK, 0, 1);
        }
        __syncthreads();

        FragU a0, a1, b0, b1;
        a0.h[0] = *(const v8bf*)(arow0 + hiHalf * 8);
        a0.h[1] = *(const v8bf*)(arow0 + 16 + hiHalf * 8);
        a1.h[0] = *(const v8bf*)(arow1 + hiHalf * 8);
        a1.h[1] = *(const v8bf*)(arow1 + 16 + hiHalf * 8);
        b0.h[0] = *(const v8bf*)(wrow0);
        b0.h[1] = *(const v8bf*)(wrow0 + 16);
        b1.h[0] = *(const v8bf*)(wrow1);
        b1.h[1] = *(const v8bf*)(wrow1 + 16);

        acc00 = __builtin_amdgcn_wmma_f32_16x16x32_bf16(false, a0.v, false, b0.v, (short)0, acc00, false, false);
        acc01 = __builtin_amdgcn_wmma_f32_16x16x32_bf16(false, a0.v, false, b1.v, (short)0, acc01, false, false);
        acc10 = __builtin_amdgcn_wmma_f32_16x16x32_bf16(false, a1.v, false, b0.v, (short)0, acc10, false, false);
        acc11 = __builtin_amdgcn_wmma_f32_16x16x32_bf16(false, a1.v, false, b1.v, (short)0, acc11, false, false);
        __syncthreads();
    }

    // Epilogue. C/D layout: VGPR r -> M = tile*16 + 8*hiHalf + r, N = laneM.
    const int n0 = bn0 + (tc0 + 0) * 16 + laneM;
    const int n1 = bn0 + (tc0 + 1) * 16 + laneM;
    const float bb0 = bias ? bias[n0] : 0.f;
    const float bb1 = bias ? bias[n1] : 0.f;
    v8f* accs[2][2] = { { &acc00, &acc01 }, { &acc10, &acc11 } };
    #pragma unroll
    for (int i = 0; i < 2; ++i) {
        const int mb = bm0 + (tr0 + i) * 16 + hiHalf * 8;
        #pragma unroll
        for (int j = 0; j < 2; ++j) {
            const int n = j ? n1 : n0;
            const float bb = j ? bb1 : bb0;
            #pragma unroll
            for (int r = 0; r < 8; ++r) {
                float v = (*accs[i][j])[r] + bb;
                if (act == 1)  // branchless stable softplus
                    v = fmaxf(v, 0.f) + log1pf(__expf(-fabsf(v)));
                C[(size_t)(mb + r) * ldc + n] = v;
            }
        }
    }
}

// Depthwise causal conv1d(k=4) + bias + SiLU, reading xs slice of xz (row stride rs)
__global__ void conv_silu_kernel(const float* __restrict__ xz,
                                 const float* __restrict__ cw,
                                 const float* __restrict__ cb,
                                 float* __restrict__ out,
                                 int Bsz, int S, int Din, int rs)
{
    size_t idx = (size_t)blockIdx.x * blockDim.x + threadIdx.x;
    size_t total = (size_t)Bsz * S * Din;
    if (idx >= total) return;
    int d = (int)(idx % Din);
    int s = (int)((idx / Din) % S);
    int b = (int)(idx / ((size_t)Din * S));
    const float* base = xz + ((size_t)b * S) * rs + d;
    float acc = cb[d];
    #pragma unroll
    for (int j = 0; j < 4; ++j) {
        int sp = s - 3 + j;
        if (sp >= 0) acc += cw[d * 4 + j] * base[(size_t)sp * rs];
    }
    out[idx] = acc / (1.f + __expf(-acc));
}

// Selective scan: one channel (b,d) per lane; h[64] + A-row in VGPRs; B/C staged
// through LDS in 16-step chunks (same-address LDS reads broadcast conflict-free).
// Fuses the D skip-connection and the z-gate: out = (y + x*D) * silu(z).
#define TS 16
__global__ __launch_bounds__(256) void scan_kernel(
    const float* __restrict__ xs,    // (B,S,2048) post conv+silu
    const float* __restrict__ dt,    // (B,S,2048) post softplus
    const float* __restrict__ dbc,   // (B,S,192): cols [64,128)=B, [128,192)=C
    const float* __restrict__ xz,    // z at col 2048+d, row stride 4096
    const float* __restrict__ A_log,
    const float* __restrict__ Dp_,
    float* __restrict__ yg,          // (B,S,2048)
    int S)
{
    __shared__ float Bs[TS][64];
    __shared__ float Cs[TS][64];
    const int tid = threadIdx.x;
    const int b = blockIdx.x >> 3;                 // D_INNER/256 = 8 groups
    const int d = (blockIdx.x & 7) * 256 + tid;

    float Arow[64];
    #pragma unroll
    for (int k = 0; k < 64; ++k) Arow[k] = -__expf(A_log[d * 64 + k]);
    float h[64];
    #pragma unroll
    for (int k = 0; k < 64; ++k) h[k] = 0.f;
    const float Dv = Dp_[d];

    for (int s0 = 0; s0 < S; s0 += TS) {
        #pragma unroll
        for (int i = 0; i < 8; ++i) {              // 2048 floats, 8/thread
            int li = tid + i * 256;
            int si = li >> 7;
            int k  = li & 127;
            float v = dbc[((size_t)b * S + (s0 + si)) * 192 + 64 + k];
            if (k < 64) Bs[si][k] = v; else Cs[si][k - 64] = v;
        }
        if (s0 + TS < S) {                         // prefetch next chunk's B/C
            __builtin_prefetch(
                &dbc[((size_t)b * S + (s0 + TS) + (tid >> 4)) * 192 + 64 + (tid & 15) * 8],
                0, 1);
        }
        __syncthreads();
        for (int si = 0; si < TS; ++si) {
            size_t row = ((size_t)b * S + (s0 + si)) * 2048 + d;
            float dtv = dt[row];
            float xv  = xs[row];
            float dx  = dtv * xv;
            float y = 0.f;
            #pragma unroll
            for (int k = 0; k < 64; ++k) {
                h[k] = h[k] * __expf(dtv * Arow[k]) + dx * Bs[si][k];
                y += h[k] * Cs[si][k];
            }
            float zv = xz[((size_t)b * S + (s0 + si)) * 4096 + 2048 + d];
            yg[row] = (y + xv * Dv) * (zv / (1.f + __expf(-zv)));
        }
        __syncthreads();
    }
}

extern "C" void kernel_launch(void* const* d_in, const int* in_sizes, int n_in,
                              void* d_out, int out_size, void* d_ws, size_t ws_size,
                              hipStream_t stream)
{
    const float* x        = (const float*)d_in[0];
    const float* in_proj  = (const float*)d_in[1];
    const float* conv_w   = (const float*)d_in[2];
    const float* conv_b   = (const float*)d_in[3];
    const float* x_proj   = (const float*)d_in[4];
    const float* dt_proj  = (const float*)d_in[5];
    const float* dt_b     = (const float*)d_in[6];
    const float* A_log    = (const float*)d_in[7];
    const float* D_param  = (const float*)d_in[8];
    const float* out_proj = (const float*)d_in[9];
    float* out = (float*)d_out;
    (void)in_sizes; (void)n_in; (void)out_size; (void)ws_size;

    const int Bsz = 4, S = 4096, Dm = 1024, Di = 2048, Rt = 64;
    const int M = Bsz * S;                         // 16384

    float* ws  = (float*)d_ws;
    float* xz  = ws;                size_t off = (size_t)M * 4096;   // (B,S,4096)
    float* xsA = ws + off;          off += (size_t)M * Di;           // conv+silu out
    float* dbc = ws + off;          off += (size_t)M * 192;          // x_proj out
    float* dtb = ws + off;          off += (size_t)M * Di;           // softplus(dt)
    float* yg  = ws + off;                                           // gated y

    dim3 blk(256);

    // 1) xz = x @ in_proj^T          (16384 x 4096, K=1024)
    wmma_gemm_bf16<<<dim3((2*Di)/64, M/128), blk, 0, stream>>>(
        x, Dm, in_proj, Dm, xz, 2*Di, nullptr, M, 2*Di, Dm, 0);

    // 2) xs = silu(causal depthwise conv(xz[:, :2048]) + b)
    {
        size_t total = (size_t)M * Di;
        conv_silu_kernel<<<(unsigned)((total + 255) / 256), blk, 0, stream>>>(
            xz, conv_w, conv_b, xsA, Bsz, S, Di, 2*Di);
    }

    // 3) dbc = xs @ x_proj^T         (16384 x 192, K=2048)
    wmma_gemm_bf16<<<dim3(192/64, M/128), blk, 0, stream>>>(
        xsA, Di, x_proj, Di, dbc, 192, nullptr, M, 192, Di, 0);

    // 4) dt = softplus(dbc[:, :64] @ dt_proj^T + dt_b)   (16384 x 2048, K=64)
    wmma_gemm_bf16<<<dim3(Di/64, M/128), blk, 0, stream>>>(
        dbc, 192, dt_proj, Rt, dtb, Di, dt_b, M, Di, Rt, 1);

    // 5) selective scan + D skip + z-gate
    scan_kernel<<<dim3(Bsz * (Di/256)), blk, 0, stream>>>(
        xsA, dtb, dbc, xz, A_log, D_param, yg, S);

    // 6) out = yg @ out_proj^T       (16384 x 1024, K=2048)
    wmma_gemm_bf16<<<dim3(Dm/64, M/128), blk, 0, stream>>>(
        yg, Di, out_proj, Di, out, Dm, nullptr, M, Dm, Di, 0);
}
